// GQAttention_53480932770686
// MI455X (gfx1250) — matrix operand order repaired and checked by reading, not compiled, not run on hardware
//
#include <hip/hip_runtime.h>
#include <hip/hip_bf16.h>
#include <math.h>
#include <stdint.h>

// ---------------------------------------------------------------------------
// CDNA5 (gfx1250) WMMA types & fragment loaders
// ---------------------------------------------------------------------------
typedef __attribute__((ext_vector_type(16))) __bf16 v16bf;
typedef __attribute__((ext_vector_type(8)))  float  v8f;

// A fragment: 16x32 (MxK) bf16 tile, LDS row-major [row][k], ld elements.
// ISA: lanes 0-15 -> M=lane, K in {0..7,16..23}; lanes 16-31 -> M=lane-16,
// K in {8..15,24..31}. Each 16-bf16 half of the fragment is one aligned 16B
// chunk -> exactly two ds_load_b128 per fragment.
__device__ __forceinline__ v16bf load_a_frag(const __bf16* base, int ld) {
  const int lane = threadIdx.x & 31;
  const int hgrp = lane >> 4;
  const int m    = lane & 15;
  union { v16bf v; uint4 q[2]; } f;
  const __bf16* rowp = base + m * ld + hgrp * 8;
  f.q[0] = *(const uint4*)(rowp);
  f.q[1] = *(const uint4*)(rowp + 16);
  return f.v;
}

// B fragment: 32x16 (KxN) bf16, LDS stored as [N][K] (K contiguous).
// ISA: lanes 0-15 -> K in {0..7,16..23}, lanes 16-31 -> K in {8..15,24..31}, N = lane & 15.
// Two aligned 8-bf16 chunks along K -> two ds_load_b128.
__device__ __forceinline__ v16bf load_b_frag(const __bf16* base, int ld) {
  const int lane = threadIdx.x & 31;
  const int kg   = lane >> 4;
  const int n    = lane & 15;
  union { v16bf v; uint4 q[2]; } f;
  const __bf16* rowp = base + n * ld + kg * 8;
  f.q[0] = *(const uint4*)(rowp);
  f.q[1] = *(const uint4*)(rowp + 16);
  return f.v;
}

#define WMMA_BF16(A, B, C) \
  __builtin_amdgcn_wmma_f32_16x16x32_bf16(false, (A), false, (B), (short)0, (C), false, false)

// ---------------------------------------------------------------------------
// CDNA5 async global->LDS copy (ASYNCcnt-tracked), 16B per lane.
// VDST = LDS byte offset (low 32 bits of generic LDS address per aperture map).
// ---------------------------------------------------------------------------
__device__ __forceinline__ void async_ldg_b128(void* lds, const void* g) {
  unsigned loff = (unsigned)(uintptr_t)lds;
  unsigned long long ga = (unsigned long long)(uintptr_t)g;
  asm volatile("global_load_async_to_lds_b128 %0, %1, off"
               :: "v"(loff), "v"(ga)
               : "memory");
}

__device__ __forceinline__ void wait_async0() {
#if __has_builtin(__builtin_amdgcn_s_wait_asynccnt)
  __builtin_amdgcn_s_wait_asynccnt(0);
#else
  asm volatile("s_wait_asynccnt 0x0" ::: "memory");
#endif
}

// ---------------------------------------------------------------------------
// Elementwise converts / transposes
// ---------------------------------------------------------------------------
__global__ __launch_bounds__(256) void cvt_f32_bf16(const float* __restrict__ x,
                                                    __bf16* __restrict__ y, int n) {
  int i = blockIdx.x * 256 + threadIdx.x;
  if (i < n) y[i] = (__bf16)x[i];
}

// W fp32 [K][N] -> Wt bf16 [N][K] (tiled via LDS, fully coalesced)
__global__ __launch_bounds__(256) void transpose_cvt(const float* __restrict__ W,
                                                     __bf16* __restrict__ Wt,
                                                     int K, int N) {
  __shared__ float tile[32][33];
  const int k0 = blockIdx.x * 32, n0 = blockIdx.y * 32;
  const int tx = threadIdx.x, ty = threadIdx.y;
#pragma unroll
  for (int i = 0; i < 32; i += 8)
    tile[ty + i][tx] = W[(size_t)(k0 + ty + i) * N + n0 + tx];
  __syncthreads();
#pragma unroll
  for (int i = 0; i < 32; i += 8)
    Wt[(size_t)(n0 + ty + i) * K + k0 + tx] = (__bf16)tile[tx][ty + i];
}

// ---------------------------------------------------------------------------
// bf16 WMMA GEMM: C[M][N](f32,+bias) = A[M][K](bf16) * Bt[N][K](bf16)^T
// BM=128 BN=64 BK=64, 256 threads (8 waves), each wave 32x32 output.
// Double-buffered LDS fed by async global->LDS copies.
// ---------------------------------------------------------------------------
__global__ __launch_bounds__(256) void gemm_bf16(const __bf16* __restrict__ A,
                                                 const __bf16* __restrict__ Bt,
                                                 float* __restrict__ C,
                                                 const float* __restrict__ bias,
                                                 int M, int N, int K) {
  __shared__ __align__(16) __bf16 As[2][128 * 72];
  __shared__ __align__(16) __bf16 Bs[2][64 * 72];
  const int tid  = threadIdx.x;
  const int lane = tid & 31;
  const int wid  = tid >> 5;
  const int m0   = blockIdx.y * 128;
  const int n0   = blockIdx.x * 64;
  const int wm   = (wid >> 1) * 32;
  const int wn   = (wid & 1) * 32;

  v8f acc[2][2] = {};

  auto tile_copy = [&](int buf, int kbase) {
#pragma unroll
    for (int c = 0; c < 4; ++c) {            // A: 128x64 bf16 = 1024 x 16B
      int flat = c * 256 + tid;
      int r = flat >> 3, s = (flat & 7) * 8;
      async_ldg_b128(&As[buf][r * 72 + s], &A[(size_t)(m0 + r) * K + kbase + s]);
    }
#pragma unroll
    for (int c = 0; c < 2; ++c) {            // B: 64x64 bf16 = 512 x 16B
      int flat = c * 256 + tid;
      int r = flat >> 3, s = (flat & 7) * 8;
      async_ldg_b128(&Bs[buf][r * 72 + s], &Bt[(size_t)(n0 + r) * K + kbase + s]);
    }
  };

  const int nkt = K / 64;
  tile_copy(0, 0);
  wait_async0();
  __syncthreads();

  int cur = 0;
  for (int kt = 0; kt < nkt; ++kt) {
    if (kt + 1 < nkt) tile_copy(cur ^ 1, (kt + 1) * 64);  // prefetch next stage

#pragma unroll
    for (int kk = 0; kk < 64; kk += 32) {
      v16bf a0 = load_a_frag(&As[cur][(wm + 0) * 72 + kk], 72);
      v16bf a1 = load_a_frag(&As[cur][(wm + 16) * 72 + kk], 72);
      v16bf b0 = load_b_frag(&Bs[cur][(wn + 0) * 72 + kk], 72);
      v16bf b1 = load_b_frag(&Bs[cur][(wn + 16) * 72 + kk], 72);
      acc[0][0] = WMMA_BF16(a0, b0, acc[0][0]);
      acc[0][1] = WMMA_BF16(a0, b1, acc[0][1]);
      acc[1][0] = WMMA_BF16(a1, b0, acc[1][0]);
      acc[1][1] = WMMA_BF16(a1, b1, acc[1][1]);
    }

    wait_async0();      // my async copies into the other buffer are done
    __syncthreads();    // everyone finished compute on `cur` + copies
    cur ^= 1;
  }

  const int hgrp = lane >> 4, l15 = lane & 15;
#pragma unroll
  for (int mi = 0; mi < 2; ++mi)
#pragma unroll
    for (int ni = 0; ni < 2; ++ni) {
      int col = n0 + wn + ni * 16 + l15;
      float bv = bias ? bias[col] : 0.0f;
#pragma unroll
      for (int r = 0; r < 8; ++r) {
        int row = m0 + wm + mi * 16 + r + 8 * hgrp;
        C[(size_t)row * N + col] = acc[mi][ni][r] + bv;
      }
    }
}

// ---------------------------------------------------------------------------
// Fused RMSNorm + RoPE per (s, head): one wave per row of 64, bf16 head-major out
// ---------------------------------------------------------------------------
__global__ __launch_bounds__(256) void rmsnorm_rope(const float* __restrict__ src,
                                                    const float* __restrict__ wscale,
                                                    const float* __restrict__ cosp,
                                                    const float* __restrict__ sinp,
                                                    __bf16* __restrict__ dst,
                                                    int S, int H) {
  const int wid  = threadIdx.x >> 5;
  const int lane = threadIdx.x & 31;
  const int flat = blockIdx.x * 8 + wid;
  const int s = flat / H;
  const int h = flat % H;
  const float* row = src + (size_t)s * H * 64 + h * 64;
  float x1 = row[lane];
  float x2 = row[lane + 32];
  float ss = x1 * x1 + x2 * x2;
#pragma unroll
  for (int m = 16; m >= 1; m >>= 1) ss += __shfl_xor(ss, m, 32);
  float rn = rsqrtf(ss * (1.0f / 64.0f) + 1e-6f);
  float n1 = x1 * rn * wscale[lane];
  float n2 = x2 * rn * wscale[lane + 32];
  float c1 = cosp[s * 64 + lane],      s1 = sinp[s * 64 + lane];
  float c2 = cosp[s * 64 + lane + 32], s2 = sinp[s * 64 + lane + 32];
  float o1 = n1 * c1 - n2 * s1;  // rot = [-x2, x1]
  float o2 = n2 * c2 + n1 * s2;
  __bf16* d = dst + ((size_t)h * S + s) * 64;
  d[lane]      = (__bf16)o1;
  d[lane + 32] = (__bf16)o2;
}

// V: fp32 [S][8*64] -> bf16 head-major [8][S][64]
__global__ __launch_bounds__(256) void permute_cvt_v(const float* __restrict__ Vf,
                                                     __bf16* __restrict__ Vh, int S) {
  int idx = blockIdx.x * 256 + threadIdx.x;  // < S*8*64
  int d = idx & 63;
  int h = (idx >> 6) & 7;
  int s = idx >> 9;
  Vh[((size_t)h * S + s) * 64 + d] = (__bf16)Vf[idx];
}

// ---------------------------------------------------------------------------
// Flash attention (causal, GQA 4:1). Block = 128 threads (4 waves), Br=Bc=64.
// K tiles: async global->LDS, double-buffered. V tiles: software-pipelined
// register-staged transpose into the idle buffer. Online softmax per row via
// wave32 half-group shuffles; P re-laid out through LDS (C-layout -> A-layout).
// ---------------------------------------------------------------------------
__global__ __launch_bounds__(128) void flash_attn(const __bf16* __restrict__ Qh,  // [32][S][64]
                                                  const __bf16* __restrict__ Kh,  // [8][S][64]
                                                  const __bf16* __restrict__ Vh,  // [8][S][64]
                                                  __bf16* __restrict__ Ob,        // [S][2048]
                                                  int S) {
  const int qt   = blockIdx.x;   // q tile (64 rows)
  const int h    = blockIdx.y;   // query head
  const int kh   = h >> 2;       // kv head (GS=4)
  const int tid  = threadIdx.x;
  const int lane = tid & 31;
  const int w    = tid >> 5;
  const int hgrp = lane >> 4;
  const int l15  = lane & 15;

  __shared__ __align__(16) __bf16 Qs[64 * 72];
  __shared__ __align__(16) __bf16 Ks[2][64 * 72];
  __shared__ __align__(16) __bf16 Vt[2][64 * 72];   // [d][key]
  __shared__ __align__(16) __bf16 Ps[64 * 72];

  uint4 tv[4];  // staged V chunk: this thread owns V row (tid>>1), d-seg (tid&1)*32

  auto kload = [&](int buf, int j) {
    const __bf16* Kg = Kh + ((size_t)kh * S + (size_t)j * 64) * 64;
#pragma unroll
    for (int c = 0; c < 4; ++c) {
      int flat = c * 128 + tid;
      int r = flat >> 3, s2 = (flat & 7) * 8;
      async_ldg_b128(&Ks[buf][r * 72 + s2], &Kg[(size_t)r * 64 + s2]);
    }
  };
  auto vload = [&](int j) {
    const __bf16* Vg = Vh + ((size_t)kh * S + (size_t)j * 64) * 64;
    const __bf16* src = Vg + (size_t)(tid >> 1) * 64 + (tid & 1) * 32;
#pragma unroll
    for (int c = 0; c < 4; ++c) tv[c] = *(const uint4*)(src + c * 8);
  };
  auto vstore = [&](int buf) {
    int k = tid >> 1, dbase = (tid & 1) * 32;
    const __bf16* e = (const __bf16*)tv;
#pragma unroll
    for (int i = 0; i < 32; ++i) Vt[buf][(dbase + i) * 72 + k] = e[i];
  };

  // prologue: Q (async) + K/V tile 0
  const __bf16* Qg = Qh + ((size_t)h * S + (size_t)qt * 64) * 64;
#pragma unroll
  for (int c = 0; c < 4; ++c) {
    int flat = c * 128 + tid;
    int r = flat >> 3, s2 = (flat & 7) * 8;
    async_ldg_b128(&Qs[r * 72 + s2], &Qg[(size_t)r * 64 + s2]);
  }
  kload(0, 0);
  vload(0);
  vstore(0);
  wait_async0();
  __syncthreads();

  float mrun[8], lrun[8];
  v8f o[4] = {};
#pragma unroll
  for (int r = 0; r < 8; ++r) { mrun[r] = -INFINITY; lrun[r] = 0.0f; }
  const float scale = 0.125f;  // 1/sqrt(64)

  int cur = 0;
  for (int j = 0; j <= qt; ++j) {
    const bool more = (j + 1 <= qt);
    if (more) { kload(cur ^ 1, j + 1); vload(j + 1); }  // overlap with compute

    // S = Q @ K^T   (A: Q rows; B: K tile already [key][d] == [N][K])
    v8f sc[4] = {};
#pragma unroll
    for (int kk = 0; kk < 64; kk += 32) {
      v16bf a = load_a_frag(&Qs[(w * 16) * 72 + kk], 72);
#pragma unroll
      for (int t = 0; t < 4; ++t) {
        v16bf b = load_b_frag(&Ks[cur][(t * 16) * 72 + kk], 72);
        sc[t] = WMMA_BF16(a, b, sc[t]);
      }
    }
    // scale + causal mask on the diagonal tile
#pragma unroll
    for (int t = 0; t < 4; ++t)
#pragma unroll
      for (int r = 0; r < 8; ++r) {
        float v = sc[t][r] * scale;
        if (j == qt) {
          int col = t * 16 + l15;
          int row = w * 16 + r + 8 * hgrp;
          if (col > row) v = -INFINITY;
        }
        sc[t][r] = v;
      }

    // online softmax (row stats live in the 16-lane half-group)
    float mnew[8];
#pragma unroll
    for (int r = 0; r < 8; ++r) {
      float m = sc[0][r];
#pragma unroll
      for (int t = 1; t < 4; ++t) m = fmaxf(m, sc[t][r]);
#pragma unroll
      for (int msk = 8; msk >= 1; msk >>= 1) m = fmaxf(m, __shfl_xor(m, msk, 32));
      mnew[r] = fmaxf(mrun[r], m);
    }
    float alpha[8];
#pragma unroll
    for (int r = 0; r < 8; ++r) { alpha[r] = __expf(mrun[r] - mnew[r]); mrun[r] = mnew[r]; }
#pragma unroll
    for (int t = 0; t < 4; ++t)
#pragma unroll
      for (int r = 0; r < 8; ++r) sc[t][r] = __expf(sc[t][r] - mrun[r]);
#pragma unroll
    for (int r = 0; r < 8; ++r) {
      float rs = sc[0][r] + sc[1][r] + sc[2][r] + sc[3][r];
#pragma unroll
      for (int msk = 8; msk >= 1; msk >>= 1) rs += __shfl_xor(rs, msk, 32);
      lrun[r] = lrun[r] * alpha[r] + rs;
    }
#pragma unroll
    for (int t = 0; t < 4; ++t)
#pragma unroll
      for (int r = 0; r < 8; ++r) o[t][r] *= alpha[r];

    // P: C-layout -> LDS A-layout (wave-private rows; LDS in-order per wave)
#pragma unroll
    for (int t = 0; t < 4; ++t)
#pragma unroll
      for (int r = 0; r < 8; ++r)
        Ps[(w * 16 + r + 8 * hgrp) * 72 + t * 16 + l15] = (__bf16)sc[t][r];

    // O += P @ V   (B: Vt = [d][key] == [N][K])
#pragma unroll
    for (int kk = 0; kk < 64; kk += 32) {
      v16bf a = load_a_frag(&Ps[(w * 16) * 72 + kk], 72);
#pragma unroll
      for (int t = 0; t < 4; ++t) {
        v16bf b = load_b_frag(&Vt[cur][(t * 16) * 72 + kk], 72);
        o[t] = WMMA_BF16(a, b, o[t]);
      }
    }

    if (more) vstore(cur ^ 1);  // drain staged V into idle buffer
    wait_async0();
    __syncthreads();
    cur ^= 1;
  }

  // normalize and store bf16 into [S][32*64] for the output projection
#pragma unroll
  for (int t = 0; t < 4; ++t) {
    int col = h * 64 + t * 16 + l15;
#pragma unroll
    for (int r = 0; r < 8; ++r) {
      int row = qt * 64 + w * 16 + r + 8 * hgrp;
      Ob[(size_t)row * 2048 + col] = (__bf16)(o[t][r] / lrun[r]);
    }
  }
}

// ---------------------------------------------------------------------------
// Host-side orchestration
// ---------------------------------------------------------------------------
extern "C" void kernel_launch(void* const* d_in, const int* in_sizes, int n_in,
                              void* d_out, int out_size, void* d_ws, size_t ws_size,
                              hipStream_t stream) {
  const float* x    = (const float*)d_in[0];
  const float* sinp = (const float*)d_in[1];
  const float* cosp = (const float*)d_in[2];
  /* d_in[3] = mask: causal, handled analytically */
  const float* Wq = (const float*)d_in[4];
  const float* Wk = (const float*)d_in[5];
  const float* Wv = (const float*)d_in[6];
  const float* Wo = (const float*)d_in[7];
  const float* bo = (const float*)d_in[8];
  const float* qs = (const float*)d_in[9];
  const float* ks = (const float*)d_in[10];
  float* out = (float*)d_out;

  const int S = 2048, D = 2048, HQ = 32, HK = 8, HD = 64;
  const int DKV = HK * HD;  // 512

  char* p = (char*)d_ws;
  auto take = [&](size_t bytes) -> void* {
    void* r = (void*)p;
    p += (bytes + 255) & ~(size_t)255;
    return r;
  };
  __bf16* xb  = (__bf16*)take((size_t)S * D * 2);
  __bf16* Wqt = (__bf16*)take((size_t)D * D * 2);      // [N=2048][K=2048]
  __bf16* Wkt = (__bf16*)take((size_t)DKV * D * 2);    // [512][2048]
  __bf16* Wvt = (__bf16*)take((size_t)DKV * D * 2);
  __bf16* Wot = (__bf16*)take((size_t)D * D * 2);
  float*  Qf  = (float*)take((size_t)S * D * 4);
  float*  Kf  = (float*)take((size_t)S * DKV * 4);
  float*  Vf  = (float*)take((size_t)S * DKV * 4);
  __bf16* qb  = (__bf16*)take((size_t)HQ * S * HD * 2);
  __bf16* kb  = (__bf16*)take((size_t)HK * S * HD * 2);
  __bf16* vb  = (__bf16*)take((size_t)HK * S * HD * 2);
  __bf16* ob  = (__bf16*)take((size_t)S * D * 2);

  // 1) converts / transposes
  cvt_f32_bf16<<<(S * D) / 256, 256, 0, stream>>>(x, xb, S * D);
  {
    dim3 blk(32, 8);
    transpose_cvt<<<dim3(D / 32, D / 32), blk, 0, stream>>>(Wq, Wqt, D, D);
    transpose_cvt<<<dim3(D / 32, DKV / 32), blk, 0, stream>>>(Wk, Wkt, D, DKV);
    transpose_cvt<<<dim3(D / 32, DKV / 32), blk, 0, stream>>>(Wv, Wvt, D, DKV);
    transpose_cvt<<<dim3(D / 32, D / 32), blk, 0, stream>>>(Wo, Wot, D, D);
  }

  // 2) QKV projections (bf16 WMMA, fp32 accumulate)
  gemm_bf16<<<dim3(D / 64, S / 128), 256, 0, stream>>>(xb, Wqt, Qf, nullptr, S, D, D);
  gemm_bf16<<<dim3(DKV / 64, S / 128), 256, 0, stream>>>(xb, Wkt, Kf, nullptr, S, DKV, D);
  gemm_bf16<<<dim3(DKV / 64, S / 128), 256, 0, stream>>>(xb, Wvt, Vf, nullptr, S, DKV, D);

  // 3) RMSNorm + RoPE -> bf16 head-major; V permute/convert
  rmsnorm_rope<<<(S * HQ) / 8, 256, 0, stream>>>(Qf, qs, cosp, sinp, qb, S, HQ);
  rmsnorm_rope<<<(S * HK) / 8, 256, 0, stream>>>(Kf, ks, cosp, sinp, kb, S, HK);
  permute_cvt_v<<<(S * HK * HD) / 256, 256, 0, stream>>>(Vf, vb, S);

  // 4) causal GQA flash attention
  flash_attn<<<dim3(S / 64, HQ), 128, 0, stream>>>(qb, kb, vb, ob, S);

  // 5) output projection + bias
  gemm_bf16<<<dim3(D / 64, S / 128), 256, 0, stream>>>(ob, Wot, out, bo, S, D, D);
}
